// NERLSTM_CRF_5471788335646
// MI455X (gfx1250) — matrix operand order repaired and checked by reading, not compiled, not run on hardware
//
#include <hip/hip_runtime.h>
#include <math.h>

typedef unsigned short u16;
typedef unsigned int u32;
typedef __attribute__((ext_vector_type(16))) __bf16 v16bf;
typedef __attribute__((ext_vector_type(8)))  float  v8f;

union Frag16 { v16bf v; uint4 q[2]; };

__device__ __forceinline__ u16 f2bf(float f) {
  u32 u = __float_as_uint(f);
  u32 r = (u + 0x7fffu + ((u >> 16) & 1u)) >> 16;   // RNE
  return (u16)r;
}
__device__ __forceinline__ float bf2f(u16 h) {
  return __uint_as_float(((u32)h) << 16);
}
__device__ __forceinline__ float sigf(float x) {
  return 1.0f / (1.0f + __expf(-x));
}
__device__ __forceinline__ v8f wmma_bf16(const Frag16& a, const Frag16& b, v8f c) {
  return __builtin_amdgcn_wmma_f32_16x16x32_bf16(false, a.v, false, b.v, (short)0, c,
                                                 false, false);
}

// CDNA5 async DMA: global -> LDS, 16B per lane, tracked by ASYNCcnt.
__device__ __forceinline__ void async_ld_b128(u32 lds_off, const void* g) {
  asm volatile("global_load_async_to_lds_b128 %0, %1, off"
               :: "v"(lds_off), "v"((unsigned long long)(uintptr_t)g)
               : "memory");
}
__device__ __forceinline__ void wait_asynccnt0() {
  asm volatile("s_wait_asynccnt 0x0" ::: "memory");
}

// Problem constants
#define TSEQ 512
#define BSZ  64
#define EDIM 256
#define HDIM 256
#define G4   1024   // 4*H

// ---------------- workspace layout (bytes, all 16B aligned) ----------------
#define OFF_XG_F  ((size_t)0)
#define OFF_XG_B  (OFF_XG_F + (size_t)TSEQ*BSZ*G4*2)            // bf16 [T*B,1024]
#define OFF_H     (OFF_XG_B + (size_t)TSEQ*BSZ*G4*2)            // bf16 [T*B,512] concat
#define OFF_EMS   (OFF_H    + (size_t)TSEQ*BSZ*512*2)           // f32  [T*B,16]
#define OFF_BPS   (OFF_EMS  + (size_t)TSEQ*BSZ*16*4)            // i32  [T*B,16]
#define OFF_EMB   (OFF_BPS  + (size_t)TSEQ*BSZ*16*4)            // bf16 [30000,256]
#define OFF_WIHF  (OFF_EMB  + (size_t)30000*256*2)
#define OFF_WIHB  (OFF_WIHF + (size_t)G4*EDIM*2)
#define OFF_WHHF  (OFF_WIHB + (size_t)G4*EDIM*2)
#define OFF_WHHB  (OFF_WHHF + (size_t)G4*HDIM*2)
#define OFF_WOUT  (OFF_WHHB + (size_t)G4*HDIM*2)                // bf16 [16,512] padded

// ---------------- kernel 0: f32 -> bf16 conversion ----------------
__global__ void k_f32_to_bf16(const float* __restrict__ src, u16* __restrict__ dst, int n) {
  int i = blockIdx.x * blockDim.x + threadIdx.x;
  if (i < n) dst[i] = f2bf(src[i]);
}

__global__ void k_pack_wout(const float* __restrict__ w, u16* __restrict__ dst) {
  int i = blockIdx.x * blockDim.x + threadIdx.x;   // 16*512
  if (i < 16 * 512) {
    int n = i >> 9, k = i & 511;
    float v = (n < 13) ? w[n * 512 + k] : 0.0f;
    dst[i] = f2bf(v);
  }
}

// ---------------- kernel 1: embedding gather + input GEMM (both dirs) ------
// out[m,n] = sum_k emb[x][m,k] * w_ih[n,k] + (b_ih[n]+b_hh[n]),  m = t*64+b
__global__ __launch_bounds__(256) void k_input_gemm(
    const int* __restrict__ x, const u16* __restrict__ emb_bf,
    const u16* __restrict__ wih_f, const u16* __restrict__ wih_b,
    const float* __restrict__ bih_f, const float* __restrict__ bhh_f,
    const float* __restrict__ bih_b, const float* __restrict__ bhh_b,
    u16* __restrict__ xg_f, u16* __restrict__ xg_b) {
  int wglob = blockIdx.x * 8 + (threadIdx.x >> 5);
  int lane = threadIdx.x & 31;
  int half = lane >> 4;
  int r    = lane & 15;

  int dir    = wglob >> 15;        // 32768 waves per direction
  int rem    = wglob & 32767;
  int m_tile = rem >> 4;
  int n_grp  = rem & 15;           // 4 N-tiles per wave -> 64 cols

  const u16*   wih = dir ? wih_b : wih_f;
  const float* b1  = dir ? bih_b : bih_f;
  const float* b2  = dir ? bhh_b : bhh_f;
  u16*         out = dir ? xg_b  : xg_f;

  int M0 = m_tile * 16;
  int myrow = M0 + r;
  int t = myrow >> 6;
  int b = myrow & 63;
  int token = x[b * TSEQ + t];
  const u16* arow = emb_bf + (size_t)token * EDIM;

  v8f acc[4];
  for (int g = 0; g < 4; ++g)
    for (int j = 0; j < 8; ++j) acc[g][j] = 0.0f;

  for (int kk = 0; kk < 8; ++kk) {
    int K0 = kk * 32;
    Frag16 af;
    af.q[0] = *(const uint4*)(arow + K0 + half * 8);
    af.q[1] = *(const uint4*)(arow + K0 + 16 + half * 8);
    Frag16 bf_[4];
    for (int g = 0; g < 4; ++g) {               // batch all B loads, then wmma x4
      int n = n_grp * 64 + g * 16 + r;
      const u16* bp = wih + (size_t)n * EDIM + K0 + half * 8;
      bf_[g].q[0] = *(const uint4*)(bp);
      bf_[g].q[1] = *(const uint4*)(bp + 16);
    }
    for (int g = 0; g < 4; ++g) acc[g] = wmma_bf16(af, bf_[g], acc[g]);
  }
  for (int g = 0; g < 4; ++g) {
    int n = n_grp * 64 + g * 16 + r;
    float bias = b1[n] + b2[n];
    for (int j = 0; j < 8; ++j) {
      int rr = M0 + half * 8 + j;
      out[(size_t)rr * G4 + n] = f2bf(acc[g][j] + bias);
    }
  }
}

// ---------------- kernel 2: persistent bidirectional LSTM recurrence -------
// grid = 8 blocks: blockIdx.x = m_blk*2 + dir ; 512 threads = 16 waves
__global__ __launch_bounds__(512) void k_lstm(
    const u16* __restrict__ xg_f, const u16* __restrict__ xg_b,
    const u16* __restrict__ whh_f, const u16* __restrict__ whh_b,
    u16* __restrict__ h_out) {
  int dir   = blockIdx.x & 1;
  int m_blk = blockIdx.x >> 1;
  const u16* xg  = dir ? xg_b  : xg_f;
  const u16* whh = dir ? whh_b : whh_f;

  int tid  = threadIdx.x;
  int wave = tid >> 5;
  int lane = tid & 31;
  int half = lane >> 4;
  int r    = lane & 15;

  __shared__ u16   h_sh[16 * HDIM];    // 8 KB  bf16 h state (16 batch rows)
  __shared__ float c_sh[16 * HDIM];    // 16 KB f32 cell state
  __shared__ u16   xg_sh[16 * G4];     // 32 KB staged gate pre-activations

  u32 xg_sh_base = (u32)(uintptr_t)(&xg_sh[0]);  // LDS byte offset = addr[31:0]

  for (int i = tid; i < 16 * HDIM; i += 512) {
    h_sh[i] = 0;
    c_sh[i] = 0.0f;
  }
  __syncthreads();

  for (int tt = 0; tt < TSEQ; ++tt) {
    int t = dir ? (TSEQ - 1 - tt) : tt;
    size_t rowbase = (size_t)(t * BSZ + m_blk * 16);

    // Stage this step's xg tile [16 x 1024] bf16 into LDS with async DMA
    // (overlaps with the WMMA K-loop below; drained before the gate math).
    for (int cidx = 0; cidx < 4; ++cidx) {
      int chunk = tid + cidx * 512;             // 2048 x 16B chunks
      int row   = chunk >> 7;                   // 128 chunks per row
      int col8  = (chunk & 127) * 8;            // u16 index
      const u16* g = xg + (rowbase + row) * G4 + col8;
      u32 loff = xg_sh_base + (u32)(row * G4 + col8) * 2u;
      async_ld_b128(loff, g);
    }
    // Warm L2/WGP caches for the tile two steps ahead.
    int tn = dir ? (t - 2) : (t + 2);
    if (tn >= 0 && tn < TSEQ) {
      const u16* p = xg + (size_t)(tn * BSZ + m_blk * 16) * G4 + tid * 32;
      __builtin_prefetch(p, 0, 0);
    }

    // g_pre = h @ w_hh.T for the 4 gate quadrants of this wave's 16 h-cols
    v8f acc[4];
    for (int g = 0; g < 4; ++g)
      for (int j = 0; j < 8; ++j) acc[g][j] = 0.0f;

    for (int kk = 0; kk < 8; ++kk) {
      int K0 = kk * 32;
      Frag16 af;
      af.q[0] = *(const uint4*)(&h_sh[r * HDIM + K0 + half * 8]);
      af.q[1] = *(const uint4*)(&h_sh[r * HDIM + K0 + 16 + half * 8]);
      Frag16 bf_[4];
      for (int g = 0; g < 4; ++g) {
        int n = g * HDIM + wave * 16 + r;
        const u16* bp = whh + (size_t)n * HDIM + K0 + half * 8;
        bf_[g].q[0] = *(const uint4*)(bp);
        bf_[g].q[1] = *(const uint4*)(bp + 16);
      }
      for (int g = 0; g < 4; ++g) acc[g] = wmma_bf16(af, bf_[g], acc[g]);
    }

    wait_asynccnt0();    // this wave's DMA done
    __syncthreads();     // all waves: h_sh reads done + xg_sh fully populated

    // elementwise gates: i | f | g | o  (PyTorch order), xg from LDS
    int nh = wave * 16 + r;
    float hv[8];
    for (int j = 0; j < 8; ++j) {
      int lr = half * 8 + j;
      const u16* xr = &xg_sh[lr * G4];
      float gi = acc[0][j] + bf2f(xr[nh]);
      float gf = acc[1][j] + bf2f(xr[HDIM + nh]);
      float gg = acc[2][j] + bf2f(xr[2 * HDIM + nh]);
      float go = acc[3][j] + bf2f(xr[3 * HDIM + nh]);
      float cold = c_sh[lr * HDIM + nh];
      float cn = sigf(gf) * cold + sigf(gi) * tanhf(gg);
      c_sh[lr * HDIM + nh] = cn;
      hv[j] = sigf(go) * tanhf(cn);
    }
    for (int j = 0; j < 8; ++j) {
      int lr = half * 8 + j;
      u16 hb = f2bf(hv[j]);
      h_sh[lr * HDIM + nh] = hb;
      h_out[(rowbase + lr) * 512 + (size_t)dir * HDIM + nh] = hb;
    }
    __syncthreads();     // new h visible before next step's GEMM
  }
}

// ---------------- kernel 3: output projection -> emissions [T*B,16] -------
__global__ __launch_bounds__(256) void k_proj(
    const u16* __restrict__ h, const u16* __restrict__ wout_pad,
    const float* __restrict__ b_out, float* __restrict__ ems) {
  int w = blockIdx.x * 8 + (threadIdx.x >> 5);   // 2048 M-tiles
  int lane = threadIdx.x & 31;
  int half = lane >> 4;
  int r    = lane & 15;
  int M0 = w * 16;

  float bias = (r < 13) ? b_out[r] : 0.0f;
  v8f acc;
  for (int j = 0; j < 8; ++j) acc[j] = bias;

  const u16* arow = h + (size_t)(M0 + r) * 512;
  for (int kk = 0; kk < 16; ++kk) {
    int K0 = kk * 32;
    Frag16 af;
    af.q[0] = *(const uint4*)(arow + K0 + half * 8);
    af.q[1] = *(const uint4*)(arow + K0 + 16 + half * 8);
    const u16* bp = wout_pad + (size_t)r * 512 + K0 + half * 8;
    Frag16 bf_;
    bf_.q[0] = *(const uint4*)(bp);
    bf_.q[1] = *(const uint4*)(bp + 16);
    acc = wmma_bf16(af, bf_, acc);
  }
  for (int j = 0; j < 8; ++j)
    ems[(size_t)(M0 + half * 8 + j) * 16 + r] = acc[j];
}

// ---------------- kernel 4: CRF Viterbi decode + backtrack -----------------
__global__ __launch_bounds__(1024) void k_viterbi(
    const float* __restrict__ ems, const float* __restrict__ start_t,
    const float* __restrict__ end_t, const float* __restrict__ trans,
    int* __restrict__ bps, float* __restrict__ out) {
  int tid = threadIdx.x;
  int b = tid >> 4;
  int k = tid & 15;

  __shared__ float sc[BSZ][16];
  __shared__ float tr[13][16];
  if (tid < 13 * 13) tr[tid / 13][tid % 13] = trans[tid];
  if (k < 13) sc[b][k] = start_t[k] + ems[(size_t)b * 16 + k];
  else        sc[b][k] = -1e30f;
  __syncthreads();

  for (int t = 1; t < TSEQ; ++t) {
    float best = -1e30f;
    int arg = 0;
    if (k < 13) {
      for (int kp = 0; kp < 13; ++kp) {
        float v = sc[b][kp] + tr[kp][k];
        if (v > best) { best = v; arg = kp; }
      }
      best += ems[((size_t)t * BSZ + b) * 16 + k];
    }
    bps[((size_t)t * BSZ + b) * 16 + k] = arg;
    __syncthreads();
    if (k < 13) sc[b][k] = best;
    __syncthreads();
  }

  if (k == 0) {
    float bs = -1e30f;
    int bl = 0;
    for (int kk = 0; kk < 13; ++kk) {
      float v = sc[b][kk] + end_t[kk];
      if (v > bs) { bs = v; bl = kk; }
    }
    out[BSZ * TSEQ + b] = bs;          // best_score
    int tag = bl;
    out[(size_t)b * TSEQ + (TSEQ - 1)] = (float)tag;
    for (int t = TSEQ - 2; t >= 0; --t) {
      tag = bps[((size_t)(t + 1) * BSZ + b) * 16 + tag];
      out[(size_t)b * TSEQ + t] = (float)tag;
    }
  }
}

// ---------------- host-side launch -----------------------------------------
extern "C" void kernel_launch(void* const* d_in, const int* in_sizes, int n_in,
                              void* d_out, int out_size, void* d_ws, size_t ws_size,
                              hipStream_t stream) {
  (void)in_sizes; (void)n_in; (void)out_size; (void)ws_size;

  const int*   x        = (const int*)  d_in[0];
  const float* emb      = (const float*)d_in[1];
  const float* w_ih_f   = (const float*)d_in[2];
  const float* w_hh_f   = (const float*)d_in[3];
  const float* b_ih_f   = (const float*)d_in[4];
  const float* b_hh_f   = (const float*)d_in[5];
  const float* w_ih_b   = (const float*)d_in[6];
  const float* w_hh_b   = (const float*)d_in[7];
  const float* b_ih_b   = (const float*)d_in[8];
  const float* b_hh_b   = (const float*)d_in[9];
  const float* w_out    = (const float*)d_in[10];
  const float* b_out    = (const float*)d_in[11];
  const float* start_tr = (const float*)d_in[12];
  const float* end_tr   = (const float*)d_in[13];
  const float* trans    = (const float*)d_in[14];

  char* ws = (char*)d_ws;
  u16*   xg_f     = (u16*)  (ws + OFF_XG_F);
  u16*   xg_b     = (u16*)  (ws + OFF_XG_B);
  u16*   h_buf    = (u16*)  (ws + OFF_H);
  float* ems      = (float*)(ws + OFF_EMS);
  int*   bps      = (int*)  (ws + OFF_BPS);
  u16*   emb_bf   = (u16*)  (ws + OFF_EMB);
  u16*   wih_f_bf = (u16*)  (ws + OFF_WIHF);
  u16*   wih_b_bf = (u16*)  (ws + OFF_WIHB);
  u16*   whh_f_bf = (u16*)  (ws + OFF_WHHF);
  u16*   whh_b_bf = (u16*)  (ws + OFF_WHHB);
  u16*   wout_bf  = (u16*)  (ws + OFF_WOUT);

  // 0) weight/embedding conversion to bf16
  {
    int n = 30000 * 256;
    k_f32_to_bf16<<<(n + 255) / 256, 256, 0, stream>>>(emb, emb_bf, n);
    n = G4 * EDIM;
    k_f32_to_bf16<<<(n + 255) / 256, 256, 0, stream>>>(w_ih_f, wih_f_bf, n);
    k_f32_to_bf16<<<(n + 255) / 256, 256, 0, stream>>>(w_ih_b, wih_b_bf, n);
    k_f32_to_bf16<<<(n + 255) / 256, 256, 0, stream>>>(w_hh_f, whh_f_bf, n);
    k_f32_to_bf16<<<(n + 255) / 256, 256, 0, stream>>>(w_hh_b, whh_b_bf, n);
    k_pack_wout<<<32, 256, 0, stream>>>(w_out, wout_bf);
  }

  // 1) embedding gather + input-side gate GEMM (both directions)
  k_input_gemm<<<8192, 256, 0, stream>>>(x, emb_bf, wih_f_bf, wih_b_bf,
                                         b_ih_f, b_hh_f, b_ih_b, b_hh_b,
                                         xg_f, xg_b);

  // 2) persistent bidirectional LSTM recurrence
  k_lstm<<<8, 512, 0, stream>>>(xg_f, xg_b, whh_f_bf, whh_b_bf, h_buf);

  // 3) output projection -> emissions
  k_proj<<<256, 256, 0, stream>>>(h_buf, wout_bf, b_out, ems);

  // 4) CRF Viterbi decode + backtrack
  k_viterbi<<<1, 1024, 0, stream>>>(ems, start_tr, end_tr, trans, bps,
                                    (float*)d_out);
}
